// AttentionLayer_9629316677989
// MI455X (gfx1250) — compile-verified, matrix operand order checked by reading, not run on hardware
//
#include <hip/hip_runtime.h>
#include <hip/hip_bf16.h>
#include <stdint.h>

typedef __attribute__((ext_vector_type(16))) __bf16 v16bf;
typedef __attribute__((ext_vector_type(8)))  float  v8f;

union Frag {
    uint4  q[2];
    v16bf  v;
};

__device__ __forceinline__ unsigned short f2bf(float f) {
    unsigned u = __builtin_bit_cast(unsigned, f);
    u += 0x7FFFu + ((u >> 16) & 1u);           // round-to-nearest-even
    return (unsigned short)(u >> 16);
}

// ---- VALU cross-lane reduction within 16-lane rows via DPP row_ror ----
template <int CTRL>
__device__ __forceinline__ float dpp_rorf(float x) {
    int i = __builtin_bit_cast(int, x);
    i = __builtin_amdgcn_update_dpp(i, i, CTRL, 0xf, 0xf, false);
    return __builtin_bit_cast(float, i);
}
__device__ __forceinline__ float rowmax16(float v) {
    v = fmaxf(v, dpp_rorf<0x121>(v));   // row_ror:1
    v = fmaxf(v, dpp_rorf<0x122>(v));   // row_ror:2
    v = fmaxf(v, dpp_rorf<0x124>(v));   // row_ror:4
    v = fmaxf(v, dpp_rorf<0x128>(v));   // row_ror:8
    return v;
}
__device__ __forceinline__ float rowsum16(float v) {
    v += dpp_rorf<0x121>(v);
    v += dpp_rorf<0x122>(v);
    v += dpp_rorf<0x124>(v);
    v += dpp_rorf<0x128>(v);
    return v;
}

// ---------------- conversion kernels ----------------

__global__ __launch_bounds__(256) void cvt_bf16_kernel(const float* __restrict__ in,
                                                       unsigned short* __restrict__ out, int n) {
    int i = blockIdx.x * blockDim.x + threadIdx.x;
    if (i < n) out[i] = f2bf(in[i]);
}

// W[512,512] fp32 row-major -> Wt bf16 with Wt[n*512+k] = W[k*512+n]
__global__ __launch_bounds__(256) void cvt_wT_kernel(const float* __restrict__ W,
                                                     unsigned short* __restrict__ Wt) {
    int i = blockIdx.x * blockDim.x + threadIdx.x;   // 0 .. 262143
    int k = i >> 9;
    int n = i & 511;
    Wt[n * 512 + k] = f2bf(W[i]);
}

// ---------------- WMMA GEMM:  Y = (A[M,512] @ W + bias) * out_scale ----------------
// A bf16 row-major, Wt bf16 transposed (Wt[n][k]).  32 (M) x 64 (N) tile per wave.
// MODE 0: bf16 out [M,512];  MODE 1: bf16 out as V^T [B,H,E,S];  MODE 2: f32 out [M,512]
template <int MODE>
__global__ __launch_bounds__(256) void gemm_bf16_kernel(const unsigned short* __restrict__ A,
                                                        const unsigned short* __restrict__ Wt,
                                                        const float* __restrict__ bias,
                                                        void* __restrict__ Out,
                                                        float out_scale) {
    constexpr int K = 512, N = 512;
    const int lane = threadIdx.x & 31;
    const int wave = threadIdx.x >> 5;
    const int lr   = lane & 15;
    const int base = (lane >> 4) * 8;                 // K-offset half / C-row half

    const size_t m0 = ((size_t)blockIdx.y * 8 + wave) * 32;
    const int    n0 = blockIdx.x * 64;

    v8f acc[2][4] = {};
    const unsigned short* arow0 = A + (m0 + (size_t)lr) * K;
    const unsigned short* arow1 = arow0 + (size_t)16 * K;

    for (int k0 = 0; k0 < K; k0 += 32) {
        Frag a0, a1;
        a0.q[0] = *(const uint4*)(arow0 + k0 + base);
        a0.q[1] = *(const uint4*)(arow0 + k0 + base + 16);
        a1.q[0] = *(const uint4*)(arow1 + k0 + base);
        a1.q[1] = *(const uint4*)(arow1 + k0 + base + 16);
#pragma unroll
        for (int t = 0; t < 4; ++t) {
            const unsigned short* wrow = Wt + (size_t)(n0 + t * 16 + lr) * K + k0 + base;
            Frag w;
            w.q[0] = *(const uint4*)(wrow);
            w.q[1] = *(const uint4*)(wrow + 16);
            acc[0][t] = __builtin_amdgcn_wmma_f32_16x16x32_bf16(false, a0.v, false, w.v,
                                                                (short)0, acc[0][t], false, false);
            acc[1][t] = __builtin_amdgcn_wmma_f32_16x16x32_bf16(false, a1.v, false, w.v,
                                                                (short)0, acc[1][t], false, false);
        }
    }

#pragma unroll
    for (int mi = 0; mi < 2; ++mi) {
#pragma unroll
        for (int t = 0; t < 4; ++t) {
            const int c = n0 + t * 16 + lr;
            const float bv = bias[c];
#pragma unroll
            for (int r = 0; r < 8; ++r) {
                const float  val = (acc[mi][t][r] + bv) * out_scale;
                const size_t m   = m0 + mi * 16 + r + base;
                if (MODE == 0) {
                    ((unsigned short*)Out)[m * N + c] = f2bf(val);
                } else if (MODE == 1) {
                    const unsigned bb = (unsigned)(m >> 11);     // S = 2048
                    const unsigned s  = (unsigned)(m & 2047);
                    const unsigned h  = (unsigned)c >> 6;        // E = 64
                    const unsigned e  = (unsigned)c & 63;
                    ((unsigned short*)Out)[(((size_t)(bb * 8u + h)) * 64 + e) * 2048 + s] = f2bf(val);
                } else {
                    ((float*)Out)[m * N + c] = val;
                }
            }
        }
    }
}

// ---------------- flash attention ----------------
// Block = 8 waves sharing one (b,h); each wave owns 16 query rows (128 rows/block).
// K/V 64-key chunks double-buffered through LDS (8x L2-traffic reduction).
// Qm (pre-scaled by 0.125*log2e), Km bf16 [B*2048,512]; Vt bf16 [B,H,64,2048]; AO bf16 [B*2048,512]
__global__ __launch_bounds__(256) void attn_kernel(const unsigned short* __restrict__ Qm,
                                                   const unsigned short* __restrict__ Km,
                                                   const unsigned short* __restrict__ Vt,
                                                   unsigned short* __restrict__ AO) {
    constexpr int Sdim = 2048, Dm = 512, Hh = 8, Eh = 64, Ld = 2048;
    constexpr int RP = 72;                       // padded row stride (halves): conflict-free
    __shared__ __align__(16) unsigned short Kbuf[2][64 * RP];   // [s][e], 9 KB each
    __shared__ __align__(16) unsigned short Vbuf[2][64 * RP];   // [e][s], 9 KB each
    __shared__ __align__(16) unsigned short Plds[8][16 * RP];   // per-wave P tile 16x64 (padded)

    const int tid  = threadIdx.x;
    const int lane = tid & 31;
    const int wave = tid >> 5;
    const int lr   = lane & 15;
    const int base = (lane >> 4) * 8;

    const int lchunk = blockIdx.x & 15;          // L / 128
    const int h      = (blockIdx.x >> 4) & 7;
    const int b      = blockIdx.x >> 7;
    const int l0     = (lchunk * 8 + wave) * 16;

    const unsigned short* Kb = Km + (size_t)b * Sdim * Dm + h * Eh;
    const unsigned short* Vb = Vt + ((size_t)(b * Hh + h)) * Eh * Sdim;

    // Q A-fragments for e-chunks [0,32) and [32,64)
    Frag aQ[2];
    const unsigned short* qrow = Qm + ((size_t)(b * Ld + l0 + lr)) * Dm + h * Eh;
#pragma unroll
    for (int e = 0; e < 2; ++e) {
        aQ[e].q[0] = *(const uint4*)(qrow + e * 32 + base);
        aQ[e].q[1] = *(const uint4*)(qrow + e * 32 + base + 16);
    }

    v8f O[4] = {};
    float mrow[8], lrow[8];
#pragma unroll
    for (int r = 0; r < 8; ++r) { mrow[r] = -3.0e38f; lrow[r] = 0.0f; }

    // cooperative stage of chunk 0:  64 rows x 64 halves = 512 x 16B, 2 x b128 per thread
    {
#pragma unroll
        for (int j = 0; j < 2; ++j) {
            const int c = j * 256 + tid;
            const int row = c >> 3, col = (c & 7) * 8;
            uint4 kv = *(const uint4*)(Kb + (size_t)row * Dm + col);
            uint4 vv = *(const uint4*)(Vb + (size_t)row * Sdim + col);
            *(uint4*)(&Kbuf[0][row * RP + col]) = kv;
            *(uint4*)(&Vbuf[0][row * RP + col]) = vv;
        }
    }
    __syncthreads();

    int cur = 0;
    for (int s0 = 0; s0 < Sdim; s0 += 64) {
        const bool have_next = (s0 + 64 < Sdim);
        const int  nxt = cur ^ 1;

        // issue global loads for next chunk early (latency hidden under compute)
        uint4 kreg[2], vreg[2];
        if (have_next) {
#pragma unroll
            for (int j = 0; j < 2; ++j) {
                const int c = j * 256 + tid;
                const int row = c >> 3, col = (c & 7) * 8;
                kreg[j] = *(const uint4*)(Kb + (size_t)(s0 + 64 + row) * Dm + col);
                vreg[j] = *(const uint4*)(Vb + (size_t)row * Sdim + s0 + 64 + col);
            }
        }

        const unsigned short* Kl = &Kbuf[cur][0];
        const unsigned short* Vl = &Vbuf[cur][0];

        // ---- scores: four 16x16 C tiles over this 64-key chunk (8 WMMAs) ----
        v8f c[4];
#pragma unroll
        for (int ht = 0; ht < 4; ++ht) {
            const unsigned short* krow = Kl + (ht * 16 + lr) * RP;
            Frag k0, k1;
            k0.q[0] = *(const uint4*)(krow + base);
            k0.q[1] = *(const uint4*)(krow + base + 16);
            k1.q[0] = *(const uint4*)(krow + 32 + base);
            k1.q[1] = *(const uint4*)(krow + 32 + base + 16);
            v8f z = {};
            c[ht] = __builtin_amdgcn_wmma_f32_16x16x32_bf16(false, aQ[0].v, false, k0.v,
                                                            (short)0, z, false, false);
            c[ht] = __builtin_amdgcn_wmma_f32_16x16x32_bf16(false, aQ[1].v, false, k1.v,
                                                            (short)0, c[ht], false, false);
        }

        // ---- online softmax update (exp2 domain, scale pre-folded into Q) ----
#pragma unroll
        for (int r = 0; r < 8; ++r) {
            float x0 = c[0][r], x1 = c[1][r], x2 = c[2][r], x3 = c[3][r];
            float tmax = rowmax16(fmaxf(fmaxf(x0, x1), fmaxf(x2, x3)));
            const float mnew = fmaxf(mrow[r], tmax);
            const float p0 = exp2f(x0 - mnew);
            const float p1 = exp2f(x1 - mnew);
            const float p2 = exp2f(x2 - mnew);
            const float p3 = exp2f(x3 - mnew);
            const float rs = rowsum16((p0 + p1) + (p2 + p3));
            const float alpha = exp2f(mrow[r] - mnew);
            mrow[r] = mnew;
            lrow[r] = lrow[r] * alpha + rs;
#pragma unroll
            for (int t = 0; t < 4; ++t) O[t][r] *= alpha;
            const int prow = r + base;
            Plds[wave][prow * RP + lr]      = f2bf(p0);
            Plds[wave][prow * RP + 16 + lr] = f2bf(p1);
            Plds[wave][prow * RP + 32 + lr] = f2bf(p2);
            Plds[wave][prow * RP + 48 + lr] = f2bf(p3);
        }
        asm volatile("s_wait_dscnt 0" ::: "memory");

        // ---- O += P @ V over two 32-wide K chunks (8 WMMAs) ----
#pragma unroll
        for (int kc = 0; kc < 2; ++kc) {
            Frag aP;
            aP.q[0] = *(const uint4*)(&Plds[wave][lr * RP + kc * 32 + base]);
            aP.q[1] = *(const uint4*)(&Plds[wave][lr * RP + kc * 32 + base + 16]);
#pragma unroll
            for (int t = 0; t < 4; ++t) {
                const unsigned short* vcol = Vl + (t * 16 + lr) * RP + kc * 32 + base;
                Frag vf;
                vf.q[0] = *(const uint4*)(vcol);
                vf.q[1] = *(const uint4*)(vcol + 16);
                O[t] = __builtin_amdgcn_wmma_f32_16x16x32_bf16(false, aP.v, false, vf.v,
                                                               (short)0, O[t], false, false);
            }
        }

        // ---- publish next chunk to LDS ----
        if (have_next) {
#pragma unroll
            for (int j = 0; j < 2; ++j) {
                const int c = j * 256 + tid;
                const int row = c >> 3, col = (c & 7) * 8;
                *(uint4*)(&Kbuf[nxt][row * RP + col]) = kreg[j];
                *(uint4*)(&Vbuf[nxt][row * RP + col]) = vreg[j];
            }
        }
        __syncthreads();
        cur = nxt;
    }

    // ---- epilogue: normalize and store bf16 [M,512] ----
#pragma unroll
    for (int r = 0; r < 8; ++r) {
        const float inv = 1.0f / lrow[r];
        const size_t row = (size_t)(b * Ld + l0 + r + base);
#pragma unroll
        for (int t = 0; t < 4; ++t)
            AO[row * Dm + h * Eh + t * 16 + lr] = f2bf(O[t][r] * inv);
    }
}

// ---------------- launcher ----------------

extern "C" void kernel_launch(void* const* d_in, const int* in_sizes, int n_in,
                              void* d_out, int out_size, void* d_ws, size_t ws_size,
                              hipStream_t stream) {
    constexpr int AEL = 4 * 2048 * 512;   // activation elements per tensor
    constexpr int WEL = 512 * 512;

    const float* q_f = (const float*)d_in[0];
    const float* k_f = (const float*)d_in[1];
    const float* v_f = (const float*)d_in[2];
    const float* Wq  = (const float*)d_in[3];
    const float* bq  = (const float*)d_in[4];
    const float* Wk  = (const float*)d_in[5];
    const float* bk  = (const float*)d_in[6];
    const float* Wv  = (const float*)d_in[7];
    const float* bv  = (const float*)d_in[8];
    const float* Wo  = (const float*)d_in[9];
    const float* bo  = (const float*)d_in[10];

    char* ws = (char*)d_ws;
    size_t off = 0;
    auto alloc = [&](size_t bytes) { char* p = ws + off; off += (bytes + 255) & ~(size_t)255; return p; };

    unsigned short* qb  = (unsigned short*)alloc((size_t)AEL * 2);
    unsigned short* kb  = (unsigned short*)alloc((size_t)AEL * 2);
    unsigned short* vb  = (unsigned short*)alloc((size_t)AEL * 2);
    unsigned short* Wqt = (unsigned short*)alloc((size_t)WEL * 2);
    unsigned short* Wkt = (unsigned short*)alloc((size_t)WEL * 2);
    unsigned short* Wvt = (unsigned short*)alloc((size_t)WEL * 2);
    unsigned short* Wot = (unsigned short*)alloc((size_t)WEL * 2);
    unsigned short* Qm  = (unsigned short*)alloc((size_t)AEL * 2);
    unsigned short* Km  = (unsigned short*)alloc((size_t)AEL * 2);
    unsigned short* Vt  = (unsigned short*)alloc((size_t)AEL * 2);
    unsigned short* AO  = (unsigned short*)alloc((size_t)AEL * 2);

    // 1) precision conversion
    cvt_bf16_kernel<<<(AEL + 255) / 256, 256, 0, stream>>>(q_f, qb, AEL);
    cvt_bf16_kernel<<<(AEL + 255) / 256, 256, 0, stream>>>(k_f, kb, AEL);
    cvt_bf16_kernel<<<(AEL + 255) / 256, 256, 0, stream>>>(v_f, vb, AEL);
    cvt_wT_kernel<<<WEL / 256, 256, 0, stream>>>(Wq, Wqt);
    cvt_wT_kernel<<<WEL / 256, 256, 0, stream>>>(Wk, Wkt);
    cvt_wT_kernel<<<WEL / 256, 256, 0, stream>>>(Wv, Wvt);
    cvt_wT_kernel<<<WEL / 256, 256, 0, stream>>>(Wo, Wot);

    // 2) QKV projections (M=8192, N=512, K=512); 32x64 tile/wave -> grid (N/64, M/256)
    dim3 ggrid(8, 32);
    const float qscale = 0.125f * 1.44269504088896340f;   // 1/sqrt(64) * log2(e), folded into Q
    gemm_bf16_kernel<0><<<ggrid, 256, 0, stream>>>(qb, Wqt, bq, Qm, qscale);
    gemm_bf16_kernel<0><<<ggrid, 256, 0, stream>>>(kb, Wkt, bk, Km, 1.0f);
    gemm_bf16_kernel<1><<<ggrid, 256, 0, stream>>>(vb, Wvt, bv, Vt, 1.0f);   // stores V^T [B,H,E,S]

    // 3) flash attention: 512 blocks x 8 waves, block shares (b,h) + LDS-staged K/V
    attn_kernel<<<512, 256, 0, stream>>>(Qm, Km, Vt, AO);

    // 4) output projection, fp32 epilogue
    gemm_bf16_kernel<2><<<ggrid, 256, 0, stream>>>(AO, Wot, bo, d_out, 1.0f);
}
